// InteractionNet_46548855554172
// MI455X (gfx1250) — matrix operand order repaired and checked by reading, not compiled, not run on hardware
//
#include <hip/hip_runtime.h>

#define NN  20000
#define EE  640000
#define FF  128
#define NBB 20

typedef __attribute__((ext_vector_type(16))) __bf16 v16bf;
typedef __attribute__((ext_vector_type(8)))  float  v8f;

union BfVec { v16bf v; unsigned int u[8]; };

__device__ __forceinline__ unsigned int f32_to_bf16_bits(float x) {
  unsigned int u = __float_as_uint(x);
  return (u + 0x7FFFu + ((u >> 16) & 1u)) >> 16;   // round-to-nearest-even
}
__device__ __forceinline__ unsigned int pack_bf16(float lo, float hi) {
  return (f32_to_bf16_bits(lo) & 0xFFFFu) | (f32_to_bf16_bits(hi) << 16);
}
// silu via v_rcp_f32 (TRANS op, co-executes) instead of IEEE divide sequence.
__device__ __forceinline__ float silu_f(float x) {
  return x * __builtin_amdgcn_rcpf(1.0f + __expf(-x));
}
// Hardware f32 atomic add (global_atomic_add_f32), no CAS loop.
__device__ __forceinline__ void atom_add_f32(float* p, float v) {
  unsafeAtomicAdd(p, v);
}

// ---------------------------------------------------------------------------
// C += A(16x128, bf16 pairs in LDS) * W(128x128, pre-fragmented bf16)
// Wave `wave` (0..7) owns output columns [wave*16, wave*16+16).
// A-fragment layout (16-bit A 16x32): lane<16 -> K {0..7,16..23}+32s,
// lane>=16 -> K {8..15,24..31}+32s, 2 bf16 per dword.
// W fragment is stored so each lane's 8 dwords are contiguous (2x b128 load).
// ---------------------------------------------------------------------------
__device__ __forceinline__ v8f gemm_tile(const unsigned int* ldsA,
                                         const unsigned int* wfrag,
                                         int wave, int lane, v8f c) {
  const int row = lane & 15;
  const int hi  = lane >> 4;
  const unsigned int* wf = wfrag + ((wave * 4) * 32 + lane) * 8;
#pragma unroll
  for (int s = 0; s < 4; ++s) {
    BfVec a, b;
    const int ka = s * 16 + hi * 4;        // dword index within 64-dword row
#pragma unroll
    for (int j = 0; j < 4; ++j) {
      a.u[j]     = ldsA[row * 64 + ka + j];
      a.u[4 + j] = ldsA[row * 64 + ka + 8 + j];
    }
#pragma unroll
    for (int j = 0; j < 8; ++j) b.u[j] = wf[j];
    wf += 256;                              // next k-step block (32 lanes * 8 dwords)
    c = __builtin_amdgcn_wmma_f32_16x16x32_bf16(false, a.v, false, b.v,
                                                (short)0, c, false, false);
  }
  return c;
}

// ---------------------------------------------------------------------------
// Pre-fragment one 128x128 f32 weight matrix into bf16 WMMA-B layout.
// dst[((chunk*4+s)*32+lane)*8+j] = pack(W[kb][col], W[kb+1][col])
// col = chunk*16+(lane&15), kb = 32*s + (lane>>4)*8 + 2j + (j>>2)*8 (B 32x16 layout).
// ---------------------------------------------------------------------------
__global__ __launch_bounds__(256) void prep_frag_kernel(const float* __restrict__ W,
                                                        unsigned int* __restrict__ dst) {
  int tid = blockIdx.x * 256 + threadIdx.x;
  if (tid >= 8192) return;
  int j     = tid & 7;
  int lane  = (tid >> 3) & 31;
  int s     = (tid >> 8) & 3;
  int chunk = tid >> 10;
  int col = chunk * 16 + (lane & 15);
  int kb  = 32 * s + (lane >> 4) * 8 + 2 * j + (j >> 2) * 8;
  dst[tid] = pack_bf16(W[(size_t)kb * FF + col], W[(size_t)(kb + 1) * FF + col]);
}

__global__ __launch_bounds__(256) void copy_kernel(float* __restrict__ dst,
                                                   const float* __restrict__ src,
                                                   long n) {
  long i = (long)blockIdx.x * blockDim.x + threadIdx.x;
  long stride = (long)gridDim.x * blockDim.x;
  for (; i < n; i += stride) dst[i] = src[i];
}

// ---------------------------------------------------------------------------
// mnp = silu(atom @ mn_w1 + b1) @ mn_w2 + b2        (16 nodes per block)
// ---------------------------------------------------------------------------
__global__ __launch_bounds__(256) void node_mlp_kernel(
    const float* __restrict__ atom, const float* __restrict__ b1,
    const float* __restrict__ b2,
    const unsigned int* __restrict__ fw1, const unsigned int* __restrict__ fw2,
    float* __restrict__ mnp) {
  __shared__ unsigned int s_a[16 * 64];
  __shared__ unsigned int s_h[16 * 64];
  const int T  = threadIdx.x;
  const int n0 = blockIdx.x * 16;
  {
    int row = T & 15, f0 = (T >> 4) * 8;
    const float* src = atom + (size_t)(n0 + row) * FF + f0;
#pragma unroll
    for (int p = 0; p < 4; ++p)
      s_a[row * 64 + (f0 >> 1) + p] = pack_bf16(src[2 * p], src[2 * p + 1]);
  }
  __syncthreads();
  const int wave = T >> 5, lane = T & 31, hi = lane >> 4;
  const int col  = wave * 16 + (lane & 15);
  v8f c = {};
  c = gemm_tile(s_a, fw1, wave, lane, c);
  unsigned short* h16 = (unsigned short*)s_h;
  const float bb1 = b1[col];
#pragma unroll
  for (int r = 0; r < 8; ++r) {
    int row = hi * 8 + r;
    h16[row * FF + col] = (unsigned short)f32_to_bf16_bits(silu_f(c[r] + bb1));
  }
  __syncthreads();
  v8f c2 = {};
  c2 = gemm_tile(s_h, fw2, wave, lane, c2);
  const float bb2 = b2[col];
#pragma unroll
  for (int r = 0; r < 8; ++r) {
    int row = hi * 8 + r;
    mnp[(size_t)(n0 + row) * FF + col] = c2[r] + bb2;
  }
}

// ---------------------------------------------------------------------------
// Fused edge kernel, persistent blocks: 16 edges per tile (256 thr = 8 waves).
//   message = (dist@me_w) * mnp[src] * mnp[dst]   (VALU, K=20)
//   atom_acc[src] += message                       (global_atomic_add_f32)
//   em1 = (silu(msg@W1a)@W1b) * dir                (bf16 WMMA, f32 acc)
//   em2 = (silu(msg@W2a)@W2b) * force_in[dst]
//   force_out[src] += em1+em2                      (global_atomic_add_f32)
// Both first-stage GEMMs read s_msg and run back-to-back; intermediates go to
// two separate LDS buffers so both second-stage GEMMs also run back-to-back.
// 4 barriers per tile.
// ---------------------------------------------------------------------------
__global__ __launch_bounds__(256) void edge_kernel(
    const float* __restrict__ mnp, float* __restrict__ atom_acc,
    float* __restrict__ force_out, const float* __restrict__ force_in,
    const float* __restrict__ dist_edge, const float* __restrict__ dir_edge,
    const long long* __restrict__ eidx, const float* __restrict__ me_w,
    const unsigned int* __restrict__ f11, const unsigned int* __restrict__ f12,
    const unsigned int* __restrict__ f21, const unsigned int* __restrict__ f22) {
  __shared__ float s_mew[NBB * FF];    // 10 KB, loaded once per block
  __shared__ float s_dist[16 * NBB];
  __shared__ float s_dir[16 * 3];
  __shared__ int   s_src[16], s_dst[16];
  __shared__ unsigned int s_msg[16 * 64];  // 16x128 bf16
  __shared__ unsigned int s_h1[16 * 64];
  __shared__ unsigned int s_h2[16 * 64];

  const int T = threadIdx.x;
  for (int i = T; i < NBB * FF; i += 256) s_mew[i] = me_w[i];

  const int wave = T >> 5, lane = T & 31, hi = lane >> 4;
  const int col  = wave * 16 + (lane & 15);
  unsigned short* h16a = (unsigned short*)s_h1;
  unsigned short* h16b = (unsigned short*)s_h2;

  const long ntiles = EE / 16;
  for (long tile = blockIdx.x; tile < ntiles; tile += gridDim.x) {
    const long e0 = tile * 16;
    __syncthreads();   // previous iteration done with all LDS (incl. s_mew 1st iter)

    for (int i = T; i < 16 * NBB; i += 256) s_dist[i] = dist_edge[e0 * NBB + i];
    if (T < 48) s_dir[T] = dir_edge[e0 * 3 + T];
    if (T < 16) {
      s_src[T] = (int)eidx[e0 + T];
      s_dst[T] = (int)eidx[(long)EE + e0 + T];
    }
    __syncthreads();

    // ---- message + inv_update1 scatter + bf16 staging ----
    {
      int row = T & 15, f0 = (T >> 4) * 8;
      int sn = s_src[row], dn = s_dst[row];
      const float* ms = mnp + (size_t)sn * FF;
      const float* md = mnp + (size_t)dn * FF;
      float m[8];
#pragma unroll
      for (int ff = 0; ff < 8; ++ff) {
        int f = f0 + ff;
        float ep = 0.0f;
#pragma unroll
        for (int q = 0; q < NBB; ++q) ep += s_dist[row * NBB + q] * s_mew[q * FF + f];
        float mm = ep * ms[f] * md[f];
        m[ff] = mm;
        atom_add_f32(&atom_acc[(size_t)sn * FF + f], mm);
      }
#pragma unroll
      for (int p = 0; p < 4; ++p)
        s_msg[row * 64 + (f0 >> 1) + p] = pack_bf16(m[2 * p], m[2 * p + 1]);
    }
    __syncthreads();

    // ---- first-stage GEMMs (both read s_msg), back-to-back WMMA ----
    v8f c1 = {}, c2 = {};
    c1 = gemm_tile(s_msg, f11, wave, lane, c1);
    c2 = gemm_tile(s_msg, f21, wave, lane, c2);
#pragma unroll
    for (int r = 0; r < 8; ++r) {
      int row = hi * 8 + r;
      h16a[row * FF + col] = (unsigned short)f32_to_bf16_bits(silu_f(c1[r]));
      h16b[row * FF + col] = (unsigned short)f32_to_bf16_bits(silu_f(c2[r]));
    }
    __syncthreads();

    // ---- second-stage GEMMs, back-to-back WMMA ----
    v8f em1c = {}, em2c = {};
    em1c = gemm_tile(s_h1, f12, wave, lane, em1c);
    em2c = gemm_tile(s_h2, f22, wave, lane, em2c);

    // ---- force_update scatter ----
#pragma unroll
    for (int r = 0; r < 8; ++r) {
      int row = hi * 8 + r;
      int sn = s_src[row], dn = s_dst[row];
      float e1 = em1c[r], e2 = em2c[r];
#pragma unroll
      for (int v = 0; v < 3; ++v) {
        float val = e1 * s_dir[row * 3 + v] +
                    e2 * force_in[((size_t)dn * 3 + v) * FF + col];
        atom_add_f32(&force_out[((size_t)sn * 3 + v) * FF + col], val);
      }
    }
  }
}

// ---------------------------------------------------------------------------
// inv_update2 + LayerNorm. One block (128 threads) per node.
// ---------------------------------------------------------------------------
__global__ __launch_bounds__(128) void final_kernel(
    const float* __restrict__ atom_acc, const float* __restrict__ force_new,
    const float* __restrict__ eu_w, const float* __restrict__ ln_g,
    const float* __restrict__ ln_b, float* __restrict__ atom_out) {
  __shared__ float fr[3 * FF];
  __shared__ float red[FF];
  const int n = blockIdx.x;
  const int f = threadIdx.x;
#pragma unroll
  for (int v = 0; v < 3; ++v)
    fr[v * FF + f] = force_new[((size_t)n * 3 + v) * FF + f];
  __syncthreads();

  float s = 0.0f;
#pragma unroll
  for (int v = 0; v < 3; ++v) {
    float g = 0.0f;
    for (int k = 0; k < FF; ++k) g += fr[v * FF + k] * eu_w[(size_t)k * FF + f];
    s += fr[v * FF + f] * g;
  }
  float x = atom_acc[(size_t)n * FF + f] + s;

  red[f] = x;
  __syncthreads();
  for (int off = 64; off > 0; off >>= 1) {
    if (f < off) red[f] += red[f + off];
    __syncthreads();
  }
  float mu = red[0] * (1.0f / FF);
  __syncthreads();
  red[f] = x * x;
  __syncthreads();
  for (int off = 64; off > 0; off >>= 1) {
    if (f < off) red[f] += red[f + off];
    __syncthreads();
  }
  float var = red[0] * (1.0f / FF) - mu * mu;
  atom_out[(size_t)n * FF + f] =
      (x - mu) * __frsqrt_rn(var + 1e-5f) * ln_g[f] + ln_b[f];
}

// ---------------------------------------------------------------------------
extern "C" void kernel_launch(void* const* d_in, const int* in_sizes, int n_in,
                              void* d_out, int out_size, void* d_ws, size_t ws_size,
                              hipStream_t stream) {
  const float*     atom_node  = (const float*)d_in[0];
  const float*     force_node = (const float*)d_in[1];
  const float*     disp_node  = (const float*)d_in[2];
  const float*     dir_edge   = (const float*)d_in[3];
  const float*     dist_edge  = (const float*)d_in[4];
  const long long* eidx       = (const long long*)d_in[5];
  const float* mn_w1  = (const float*)d_in[6];
  const float* mn_b1  = (const float*)d_in[7];
  const float* mn_w2  = (const float*)d_in[8];
  const float* mn_b2  = (const float*)d_in[9];
  const float* me_w   = (const float*)d_in[10];
  const float* eq1_w1 = (const float*)d_in[11];
  const float* eq1_w2 = (const float*)d_in[12];
  const float* eq2_w1 = (const float*)d_in[13];
  const float* eq2_w2 = (const float*)d_in[14];
  const float* eu_w   = (const float*)d_in[15];
  const float* ln_g   = (const float*)d_in[16];
  const float* ln_b   = (const float*)d_in[17];

  float* atom_out  = (float*)d_out;
  float* force_out = atom_out + (size_t)NN * FF;
  float* disp_out  = force_out + (size_t)NN * 3 * FF;

  float* mnp      = (float*)d_ws;                    // N*F f32
  float* atom_acc = mnp + (size_t)NN * FF;           // N*F f32
  unsigned int* frag = (unsigned int*)(atom_acc + (size_t)NN * FF);  // 6*8192 u32

  // Pre-fragment the six F x F weight matrices into bf16 WMMA-B layout.
  prep_frag_kernel<<<32, 256, 0, stream>>>(mn_w1,  frag + 0 * 8192);
  prep_frag_kernel<<<32, 256, 0, stream>>>(mn_w2,  frag + 1 * 8192);
  prep_frag_kernel<<<32, 256, 0, stream>>>(eq1_w1, frag + 2 * 8192);
  prep_frag_kernel<<<32, 256, 0, stream>>>(eq1_w2, frag + 3 * 8192);
  prep_frag_kernel<<<32, 256, 0, stream>>>(eq2_w1, frag + 4 * 8192);
  prep_frag_kernel<<<32, 256, 0, stream>>>(eq2_w2, frag + 5 * 8192);

  // Initialize accumulation targets.
  copy_kernel<<<2048, 256, 0, stream>>>(force_out, force_node, (long)NN * 3 * FF);
  copy_kernel<<<2048, 256, 0, stream>>>(disp_out,  disp_node,  (long)NN * 3 * FF);
  copy_kernel<<<1024, 256, 0, stream>>>(atom_acc,  atom_node,  (long)NN * FF);

  // Node-part MLP (WMMA).
  node_mlp_kernel<<<NN / 16, 256, 0, stream>>>(atom_node, mn_b1, mn_b2,
                                               frag + 0 * 8192, frag + 1 * 8192, mnp);

  // Fused per-edge pipeline (WMMA + scatter atomics), persistent blocks.
  edge_kernel<<<4096, 256, 0, stream>>>(mnp, atom_acc, force_out, force_node,
                                        dist_edge, dir_edge, eidx, me_w,
                                        frag + 2 * 8192, frag + 3 * 8192,
                                        frag + 4 * 8192, frag + 5 * 8192);

  // inv_update2 + LayerNorm.
  final_kernel<<<NN, 128, 0, stream>>>(atom_acc, force_out, eu_w, ln_g, ln_b, atom_out);
}
